// FdamLocalFreqEnhanceBlock_6571299963403
// MI455X (gfx1250) — matrix operand order repaired and checked by reading, not compiled, not run on hardware
//
#include <hip/hip_runtime.h>
#include <hip/hip_bf16.h>
#include <math.h>

typedef _Float16 h16 __attribute__((ext_vector_type(16)));
typedef _Float16 h8  __attribute__((ext_vector_type(8)));
typedef float    f8v __attribute__((ext_vector_type(8)));

#define B_    16
#define C_    512
#define NTOK  1024
#define HEADS 8
#define HD    64
#define QKV3  1536
#define HMLP  2048

#define WMMA_F16(a,b,c) __builtin_amdgcn_wmma_f32_16x16x32_f16(false,(a),false,(b),(short)0,(c),false,false)

// A-fragment (16x32 f16): per lane two contiguous 8-half chunks at p and p+8.
__device__ __forceinline__ h16 afrag(const _Float16* p) {
    h8 lo = *(const h8*)p;
    h8 hi = *(const h8*)(p + 8);
    return __builtin_shufflevector(lo, hi, 0,1,2,3,4,5,6,7,8,9,10,11,12,13,14,15);
}

// ---------------------------------------------------------------- conversions
__global__ __launch_bounds__(256) void cvt_f32_f16(const float* __restrict__ s,
                                                   _Float16* __restrict__ d, int n) {
    int i = blockIdx.x * 256 + threadIdx.x;
    if (i < n) d[i] = (_Float16)s[i];
}

// ---------------------------------------------------------------- layernorm
// trans=1: src is (B,C,N) read as tokens[b,n,c]=src[(b*C+c)*N+n]; trans=0: src is (B,N,C).
__global__ __launch_bounds__(256) void ln_kernel(const float* __restrict__ src,
                                                 const float* __restrict__ w,
                                                 const float* __restrict__ bia,
                                                 _Float16* __restrict__ out, int trans) {
    int tok = blockIdx.x;
    int b = tok >> 10, n = tok & 1023;
    int c0 = threadIdx.x, c1 = threadIdx.x + 256;
    size_t i0 = trans ? ((size_t)(b * C_ + c0) * NTOK + n) : ((size_t)tok * C_ + c0);
    size_t i1 = trans ? ((size_t)(b * C_ + c1) * NTOK + n) : ((size_t)tok * C_ + c1);
    float v0 = src[i0], v1 = src[i1];
    __shared__ float s1[256], s2[256];
    s1[threadIdx.x] = v0 + v1;
    s2[threadIdx.x] = v0 * v0 + v1 * v1;
    __syncthreads();
    for (int st = 128; st > 0; st >>= 1) {
        if (threadIdx.x < st) { s1[threadIdx.x] += s1[threadIdx.x + st];
                                s2[threadIdx.x] += s2[threadIdx.x + st]; }
        __syncthreads();
    }
    float mu = s1[0] * (1.f / 512.f);
    float var = s2[0] * (1.f / 512.f) - mu * mu;
    float rs = rsqrtf(var + 1e-5f);
    out[(size_t)tok * C_ + c0] = (_Float16)((v0 - mu) * rs * w[c0] + bia[c0]);
    out[(size_t)tok * C_ + c1] = (_Float16)((v1 - mu) * rs * w[c1] + bia[c1]);
}

// ---------------------------------------------------------------- generic WMMA GEMM
// Y = A(MxK, f16, row-major) * W(NdxK, f16, row-major)^T (+bias)
// mode 0: f16 plain out, 1: f16 out + exact GELU, 2: f32 out transposed to (B, Nd, NTOK)
__global__ __launch_bounds__(256) void wmma_gemm(const _Float16* __restrict__ A,
                                                 const _Float16* __restrict__ W,
                                                 void* __restrict__ out,
                                                 const float* __restrict__ bias,
                                                 int M, int Nd, int K, int mode) {
    int wave = threadIdx.x >> 5, lane = threadIdx.x & 31;
    int laneN = lane & 15, laneHi = lane >> 4;
    int mbase = blockIdx.x * 128 + wave * 16;
    int nbase = blockIdx.y * 64;
    const _Float16* Ap = A + (size_t)(mbase + laneN) * K;
    const _Float16* W0 = W + (size_t)(nbase +      laneN) * K + laneHi * 16;
    const _Float16* W1 = W + (size_t)(nbase + 16 + laneN) * K + laneHi * 16;
    const _Float16* W2 = W + (size_t)(nbase + 32 + laneN) * K + laneHi * 16;
    const _Float16* W3 = W + (size_t)(nbase + 48 + laneN) * K + laneHi * 16;
    f8v acc[4] = {};
    for (int k = 0; k < K; k += 32) {
        __builtin_prefetch((const void*)(Ap + k + 128), 0, 1);
        h16 af = afrag(Ap + k + laneHi * 16);
        h16 b0 = *(const h16*)(W0 + k);
        h16 b1 = *(const h16*)(W1 + k);
        h16 b2 = *(const h16*)(W2 + k);
        h16 b3 = *(const h16*)(W3 + k);
        acc[0] = WMMA_F16(af, b0, acc[0]);
        acc[1] = WMMA_F16(af, b1, acc[1]);
        acc[2] = WMMA_F16(af, b2, acc[2]);
        acc[3] = WMMA_F16(af, b3, acc[3]);
    }
#pragma unroll
    for (int t = 0; t < 4; t++) {
        int n = nbase + t * 16 + laneN;
        float bv = bias ? bias[n] : 0.f;
#pragma unroll
        for (int r = 0; r < 8; r++) {
            int m = mbase + r + laneHi * 8;
            float v = acc[t][r] + bv;
            if (mode == 1) v = 0.5f * v * (1.f + erff(v * 0.70710678118654752f));
            if (mode == 2) {
                int bb = m >> 10, tt = m & 1023;
                ((float*)out)[((size_t)(bb * Nd + n)) * NTOK + tt] = v;
            } else {
                ((_Float16*)out)[(size_t)m * Nd + n] = (_Float16)v;
            }
        }
    }
}

// ---------------------------------------------------------------- lf/hf gating GEMVs
__global__ __launch_bounds__(128) void lfhf_kernel(const _Float16* __restrict__ ln,
                                                   const float* __restrict__ lf_w,
                                                   const float* __restrict__ lf_b,
                                                   const float* __restrict__ hf_w,
                                                   const float* __restrict__ hf_b,
                                                   const float* __restrict__ pa_scale,
                                                   const float* __restrict__ pa_bias,
                                                   float* __restrict__ lf, float* __restrict__ hf) {
    int tok = blockIdx.x;
    float ps = pa_scale[0], pb = pa_bias[0];
    float acc[16];
#pragma unroll
    for (int o = 0; o < 16; o++) acc[o] = 0.f;
    for (int c = threadIdx.x; c < C_; c += 128) {
        float x = (float)ln[(size_t)tok * C_ + c];
        float d = (x > 0.f) ? (ps * x * x + pb) : pb;
#pragma unroll
        for (int o = 0; o < 8; o++) {
            acc[o]     += d * lf_w[o * C_ + c];
            acc[8 + o] += d * hf_w[o * C_ + c];
        }
    }
    __shared__ float sm[16][128];
#pragma unroll
    for (int o = 0; o < 16; o++) sm[o][threadIdx.x] = acc[o];
    __syncthreads();
    if (threadIdx.x < 16) {
        float s = 0.f;
        for (int i = 0; i < 128; i++) s += sm[threadIdx.x][i];
        int o = threadIdx.x;
        if (o < 8) {
            lf[(size_t)tok * 8 + o] = tanhf(s + lf_b[o]);
        } else {
            int oo = o - 8;
            float sp = log1pf(__expf(s + hf_b[oo]));
            float sq = sp * sp;
            hf[(size_t)tok * 8 + oo] = 2.f * sq / (sq + 0.3678f);
        }
    }
}

// ---------------------------------------------------------------- V transpose (bh,d,t)
__global__ __launch_bounds__(256) void vtrans_kernel(const _Float16* __restrict__ qkv,
                                                     _Float16* __restrict__ vT) {
    size_t i = (size_t)blockIdx.x * 256 + threadIdx.x; // over B*H*HD*NTOK
    int t = (int)(i & 1023);
    size_t j = i >> 10;
    int d = (int)(j & 63);
    int bh = (int)(j >> 6);
    int b = bh >> 3, h = bh & 7;
    vT[i] = qkv[((size_t)(b * NTOK + t)) * QKV3 + 2 * C_ + h * HD + d];
}

// ---------------------------------------------------------------- fused flash attention
__global__ __launch_bounds__(128) void attn_kernel(const _Float16* __restrict__ qkv,
                                                   const _Float16* __restrict__ vT,
                                                   float* __restrict__ outO) {
    __shared__ __align__(32) _Float16 pl[4][16 * 32];
    int wave = threadIdx.x >> 5, lane = threadIdx.x & 31;
    int laneN = lane & 15, laneHi = lane >> 4;
    int bh = blockIdx.y, b = bh >> 3, h = bh & 7;
    int q0 = blockIdx.x * 64 + wave * 16;

    const _Float16* qrow = qkv + ((size_t)(b * NTOK + q0 + laneN)) * QKV3 + h * HD;
    h16 qa0 = afrag(qrow + 0  + laneHi * 16);   // head-dim 0..31
    h16 qa1 = afrag(qrow + 32 + laneHi * 16);   // head-dim 32..63

    f8v accO[4] = {};
    float rmax[8], rsum[8];
#pragma unroll
    for (int r = 0; r < 8; r++) { rmax[r] = -1e30f; rsum[r] = 0.f; }

    const _Float16* kb  = qkv + ((size_t)(b * NTOK)) * QKV3 + C_ + h * HD;
    const _Float16* vtb = vT + ((size_t)bh * HD) * NTOK;
    _Float16* plw = &pl[wave][0];
    const float sc = 0.125f;  // hd^-0.5

    for (int kt = 0; kt < NTOK; kt += 32) {
        const _Float16* kr0 = kb + (size_t)(kt + laneN) * QKV3;
        const _Float16* kr1 = kb + (size_t)(kt + 16 + laneN) * QKV3;
        h16 kb00 = *(const h16*)(kr0 + 0  + laneHi * 16);
        h16 kb01 = *(const h16*)(kr0 + 32 + laneHi * 16);
        h16 kb10 = *(const h16*)(kr1 + 0  + laneHi * 16);
        h16 kb11 = *(const h16*)(kr1 + 32 + laneHi * 16);
        f8v s0 = {}, s1 = {};
        s0 = WMMA_F16(qa0, kb00, s0);
        s0 = WMMA_F16(qa1, kb01, s0);
        s1 = WMMA_F16(qa0, kb10, s1);
        s1 = WMMA_F16(qa1, kb11, s1);
#pragma unroll
        for (int r = 0; r < 8; r++) {
            float v0 = s0[r] * sc, v1 = s1[r] * sc;
            float mx = fmaxf(v0, v1);
#pragma unroll
            for (int off = 1; off < 16; off <<= 1) mx = fmaxf(mx, __shfl_xor(mx, off, 32));
            float mnew = fmaxf(rmax[r], mx);
            float p0 = __expf(v0 - mnew), p1 = __expf(v1 - mnew);
            float ps = p0 + p1;
#pragma unroll
            for (int off = 1; off < 16; off <<= 1) ps += __shfl_xor(ps, off, 32);
            float alpha = __expf(rmax[r] - mnew);
            rsum[r] = rsum[r] * alpha + ps;
            rmax[r] = mnew;
            accO[0][r] *= alpha; accO[1][r] *= alpha;
            accO[2][r] *= alpha; accO[3][r] *= alpha;
            int row = r + laneHi * 8;
            plw[row * 32 + laneN]      = (_Float16)p0;
            plw[row * 32 + 16 + laneN] = (_Float16)p1;
        }
        asm volatile("s_wait_dscnt 0x0" ::: "memory");
        h8 plo = *(const h8*)(plw + laneN * 32 + laneHi * 16);
        h8 phi = *(const h8*)(plw + laneN * 32 + 8 + laneHi * 16);
        h16 pa = __builtin_shufflevector(plo, phi, 0,1,2,3,4,5,6,7,8,9,10,11,12,13,14,15);
#pragma unroll
        for (int t = 0; t < 4; t++) {
            const _Float16* vp = vtb + (size_t)(t * 16 + laneN) * NTOK + kt + laneHi * 16;
            h16 vb = *(const h16*)vp;
            accO[t] = WMMA_F16(pa, vb, accO[t]);
        }
    }
    float* ob = outO + ((size_t)(b * NTOK + q0)) * C_ + h * HD;
#pragma unroll
    for (int r = 0; r < 8; r++) {
        float inv = 1.f / rsum[r];
        int row = r + laneHi * 8;
#pragma unroll
        for (int t = 0; t < 4; t++)
            ob[(size_t)row * C_ + t * 16 + laneN] = accO[t][r] * inv;
    }
}

// ---------------------------------------------------------------- attention combine
__global__ __launch_bounds__(256) void combine_kernel(const float* __restrict__ attn,
                                                      const _Float16* __restrict__ qkv,
                                                      const float* __restrict__ lf,
                                                      const float* __restrict__ hf,
                                                      const float* __restrict__ lfg,
                                                      const float* __restrict__ hfg,
                                                      _Float16* __restrict__ mix) {
    size_t i = (size_t)blockIdx.x * 256 + threadIdx.x; // B*N*C
    int c = (int)(i & 511);
    size_t tok = i >> 9;
    int h = c >> 6, d = c & 63;
    float a = attn[i];
    float v = (float)qkv[tok * QKV3 + 2 * C_ + h * HD + d];
    float l = lf[tok * 8 + h], f = hf[tok * 8 + h];
    float r = a + a * l * lfg[c] + f * (v - a) * hfg[c];
    mix[i] = (_Float16)r;
}

// ---------------------------------------------------------------- spatial mean pool
__global__ __launch_bounds__(128) void pool_kernel(const float* __restrict__ feat,
                                                   float* __restrict__ pooled) {
    int bc = blockIdx.x;
    float s = 0.f;
    for (int n = threadIdx.x; n < NTOK; n += 128) s += feat[(size_t)bc * NTOK + n];
    __shared__ float sm[128];
    sm[threadIdx.x] = s;
    __syncthreads();
    for (int st = 64; st > 0; st >>= 1) {
        if (threadIdx.x < st) sm[threadIdx.x] += sm[threadIdx.x + st];
        __syncthreads();
    }
    if (threadIdx.x == 0) pooled[bc] = sm[0] * (1.f / 1024.f);
}

// ---------------------------------------------------------------- routing MLP (tiny)
__global__ __launch_bounds__(64) void route_kernel(const float* __restrict__ pooled,
                                                   const float* __restrict__ rw1,
                                                   const float* __restrict__ sc,
                                                   const float* __restrict__ bi,
                                                   const float* __restrict__ rw2,
                                                   float* __restrict__ routing) {
    int b = blockIdx.x;
    __shared__ float h1[64];
    int j = threadIdx.x;
    float s = 0.f;
    for (int c = 0; c < C_; c++) s += pooled[b * C_ + c] * rw1[j * C_ + c];
    h1[j] = (s > 0.f) ? (sc[0] * s * s + bi[0]) : bi[0];
    __syncthreads();
    float o = 0.f;
    for (int k = 0; k < 64; k++) o += h1[k] * rw2[j * 64 + k];
    routing[b * 64 + j] = tanhf(o);
}

// ---------------------------------------------------------------- spectral filter
// One wave per (b,c) 32x32 plane: rfft2 (as DFT sums) * dyn * irfft2 + residual fusion.
// mode 1: out tokens2[(b*N+n)*C+c] = x[b,c,n] + gamma[c]*res   (base = x, (B,C,N))
// mode 2: out[(b*C+c)*N+n] = tokens2[b,n,c] + gamma[c]*res     (base = tokens2, (B,N,C))
__global__ __launch_bounds__(32) void freq_kernel(const float* __restrict__ feat,
                                                  const float* __restrict__ routing,
                                                  const float* __restrict__ fw,
                                                  const float* __restrict__ gamma,
                                                  const float* __restrict__ base,
                                                  float* __restrict__ outp, int mode) {
    int bc = blockIdx.x;
    int b = bc >> 9, c = bc & 511;
    int g = c >> 5, cg = c & 31;
    int lane = threadIdx.x;
    __shared__ float X[32][32];
    __shared__ float Zr[32][17], Zi[32][17];
    __shared__ float Wr[32][17], Wi[32][17];
    __shared__ float ct[32], st[32];
    float ang = (float)lane * (6.283185307179586f / 32.f);
    ct[lane] = __cosf(ang);
    st[lane] = __sinf(ang);
    const float* fp = feat + (size_t)bc * NTOK;
    for (int w = 0; w < 32; w++) X[lane][w] = fp[lane * 32 + w];
    __syncthreads();
    // forward DFT along w (row = lane)
    for (int kw = 0; kw < 17; kw++) {
        float ar = 0.f, ai = 0.f;
        for (int w = 0; w < 32; w++) {
            int idx = (w * kw) & 31;
            float xv = X[lane][w];
            ar += xv * ct[idx];
            ai -= xv * st[idx];
        }
        Zr[lane][kw] = ar; Zi[lane][kw] = ai;
    }
    __syncthreads();
    // forward DFT along h + dynamic filter (column = lane<17)
    float r0 = routing[b * 64 + g * 4 + 0];
    float r1 = routing[b * 64 + g * 4 + 1];
    float r2 = routing[b * 64 + g * 4 + 2];
    float r3 = routing[b * 64 + g * 4 + 3];
    if (lane < 17) {
        int kw = lane;
        for (int kh = 0; kh < 32; kh++) {
            float ar = 0.f, ai = 0.f;
            for (int hh = 0; hh < 32; hh++) {
                int idx = (hh * kh) & 31;
                float cc = ct[idx], ss = st[idx];
                float zr = Zr[hh][kw], zi = Zi[hh][kw];
                ar += zr * cc + zi * ss;
                ai += zi * cc - zr * ss;
            }
            size_t fo = ((size_t)cg * 32 + kh) * 17 + kw;
            const size_t FS = (size_t)32 * 32 * 17;
            float dyn = r0 * fw[fo] + r1 * fw[fo + FS] + r2 * fw[fo + 2 * FS] + r3 * fw[fo + 3 * FS];
            Wr[kh][kw] = ar * dyn; Wi[kh][kw] = ai * dyn;
        }
    }
    __syncthreads();
    // inverse DFT along h
    if (lane < 17) {
        int kw = lane;
        for (int hh = 0; hh < 32; hh++) {
            float ar = 0.f, ai = 0.f;
            for (int kh = 0; kh < 32; kh++) {
                int idx = (hh * kh) & 31;
                float cc = ct[idx], ss = st[idx];
                float wr = Wr[kh][kw], wi = Wi[kh][kw];
                ar += wr * cc - wi * ss;
                ai += wr * ss + wi * cc;
            }
            Zr[hh][kw] = ar; Zi[hh][kw] = ai;
        }
    }
    __syncthreads();
    // inverse real DFT along w (row = lane) + residual/epilogue
    const float inv = 1.f / 1024.f;  // ortho fwd * ortho inv
    for (int w = 0; w < 32; w++) {
        float acc = 0.f;
        for (int kw = 0; kw < 17; kw++) {
            int idx = (kw * w) & 31;
            float wgt = (kw == 0 || kw == 16) ? 1.f : 2.f;
            acc += wgt * (Zr[lane][kw] * ct[idx] - Zi[lane][kw] * st[idx]);
        }
        float res = acc * inv + X[lane][w];  // freq_scale(x) + x
        int n = lane * 32 + w;
        if (mode == 1) {
            outp[((size_t)(b * NTOK + n)) * C_ + c] =
                base[(size_t)bc * NTOK + n] + gamma[c] * res;
        } else {
            outp[(size_t)bc * NTOK + n] =
                base[((size_t)(b * NTOK + n)) * C_ + c] + gamma[c] * res;
        }
    }
}

// ================================================================ host
extern "C" void kernel_launch(void* const* d_in, const int* in_sizes, int n_in,
                              void* d_out, int out_size, void* d_ws, size_t ws_size,
                              hipStream_t stream) {
    const float* x        = (const float*)d_in[0];
    const float* norm1_w  = (const float*)d_in[1];
    const float* norm1_b  = (const float*)d_in[2];
    const float* qkv_w    = (const float*)d_in[3];
    const float* proj_w   = (const float*)d_in[4];
    const float* proj_b   = (const float*)d_in[5];
    const float* pa_scale = (const float*)d_in[6];
    const float* pa_bias  = (const float*)d_in[7];
    const float* lf_w     = (const float*)d_in[8];
    const float* lf_b     = (const float*)d_in[9];
    const float* hf_w     = (const float*)d_in[10];
    const float* hf_b     = (const float*)d_in[11];
    const float* lf_gamma = (const float*)d_in[12];
    const float* hf_gamma = (const float*)d_in[13];
    const float* fs1_rw1_w     = (const float*)d_in[14];
    const float* fs1_rw1_scale = (const float*)d_in[15];
    const float* fs1_rw1_bias  = (const float*)d_in[16];
    const float* fs1_rw2_w     = (const float*)d_in[17];
    const float* fs1_fw        = (const float*)d_in[18];
    const float* gamma_1  = (const float*)d_in[19];
    const float* norm2_w  = (const float*)d_in[20];
    const float* norm2_b  = (const float*)d_in[21];
    const float* fc1_w    = (const float*)d_in[22];
    const float* fc1_b    = (const float*)d_in[23];
    const float* fc2_w    = (const float*)d_in[24];
    const float* fc2_b    = (const float*)d_in[25];
    const float* fs2_rw1_w     = (const float*)d_in[26];
    const float* fs2_rw1_scale = (const float*)d_in[27];
    const float* fs2_rw1_bias  = (const float*)d_in[28];
    const float* fs2_rw2_w     = (const float*)d_in[29];
    const float* fs2_fw        = (const float*)d_in[30];
    const float* gamma_2  = (const float*)d_in[31];

    char* ws = (char*)d_ws;
    // arena (regions reused across phases)
    _Float16* bufA  = (_Float16*)(ws + 0);          // qkv f16 (48MB) then fc1 out (64MB)
    _Float16* lnb   = (_Float16*)(ws + 67108864);   // ln1 then ln2 f16 (16MB)
    float*    feat32= (float*)  (ws + 83886080);    // attn_out -> att_feat -> mlp_feat (32MB)
    _Float16* mixb  = (_Float16*)(ws + 117440512);  // vT then mix f16 (16MB)
    float*    tok2  = (float*)  (ws + 134217728);   // tokens2 f32 (32MB, live to end)
    char*     sm    = ws + 167772160;
    float*    lf    = (float*)sm;            sm += 524288;
    float*    hf    = (float*)sm;            sm += 524288;
    float*    pooled= (float*)sm;            sm += 32768;
    float*    routing=(float*)sm;            sm += 4096;
    _Float16* wq    = (_Float16*)sm;         sm += 1572864;  // 1536x512
    _Float16* wp    = (_Float16*)sm;         sm += 524288;   // 512x512
    _Float16* w1    = (_Float16*)sm;         sm += 2097152;  // 2048x512
    _Float16* w2    = (_Float16*)sm;         sm += 2097152;  // 512x2048
    float*    outF  = (float*)d_out;

    // weight conversion
    cvt_f32_f16<<<(QKV3 * C_ + 255) / 256, 256, 0, stream>>>(qkv_w, wq, QKV3 * C_);
    cvt_f32_f16<<<(C_ * C_ + 255) / 256, 256, 0, stream>>>(proj_w, wp, C_ * C_);
    cvt_f32_f16<<<(HMLP * C_ + 255) / 256, 256, 0, stream>>>(fc1_w, w1, HMLP * C_);
    cvt_f32_f16<<<(C_ * HMLP + 255) / 256, 256, 0, stream>>>(fc2_w, w2, C_ * HMLP);

    const int M = B_ * NTOK;  // 16384

    // --- attention branch ---
    ln_kernel<<<M, 256, 0, stream>>>(x, norm1_w, norm1_b, lnb, 1);
    wmma_gemm<<<dim3(M / 128, QKV3 / 64), 256, 0, stream>>>(lnb, wq, (void*)bufA, nullptr,
                                                            M, QKV3, C_, 0);
    lfhf_kernel<<<M, 128, 0, stream>>>(lnb, lf_w, lf_b, hf_w, hf_b, pa_scale, pa_bias, lf, hf);
    vtrans_kernel<<<(B_ * HEADS * HD * NTOK) / 256, 256, 0, stream>>>(bufA, mixb);
    attn_kernel<<<dim3(NTOK / 64, B_ * HEADS), 128, 0, stream>>>(bufA, mixb, feat32);
    combine_kernel<<<((size_t)M * C_) / 256, 256, 0, stream>>>(feat32, bufA, lf, hf,
                                                               lf_gamma, hf_gamma, mixb);
    wmma_gemm<<<dim3(M / 128, C_ / 64), 256, 0, stream>>>(mixb, wp, (void*)feat32, proj_b,
                                                          M, C_, C_, 2);
    // --- freq scale 1 (+ gamma_1 residual into tokens2) ---
    pool_kernel<<<B_ * C_, 128, 0, stream>>>(feat32, pooled);
    route_kernel<<<B_, 64, 0, stream>>>(pooled, fs1_rw1_w, fs1_rw1_scale, fs1_rw1_bias,
                                        fs1_rw2_w, routing);
    freq_kernel<<<B_ * C_, 32, 0, stream>>>(feat32, routing, fs1_fw, gamma_1, x, tok2, 1);

    // --- MLP branch ---
    ln_kernel<<<M, 256, 0, stream>>>(tok2, norm2_w, norm2_b, lnb, 0);
    wmma_gemm<<<dim3(M / 128, HMLP / 64), 256, 0, stream>>>(lnb, w1, (void*)bufA, fc1_b,
                                                            M, HMLP, C_, 1);
    wmma_gemm<<<dim3(M / 128, C_ / 64), 256, 0, stream>>>(bufA, w2, (void*)feat32, fc2_b,
                                                          M, C_, HMLP, 2);
    // --- freq scale 2 (+ gamma_2 residual, final (B,C,H,W) output) ---
    pool_kernel<<<B_ * C_, 128, 0, stream>>>(feat32, pooled);
    route_kernel<<<B_, 64, 0, stream>>>(pooled, fs2_rw1_w, fs2_rw1_scale, fs2_rw1_bias,
                                        fs2_rw2_w, routing);
    freq_kernel<<<B_ * C_, 32, 0, stream>>>(feat32, routing, fs2_fw, gamma_2, tok2, outF, 2);
}
